// LSTMDecoder_10256381903103
// MI455X (gfx1250) — matrix operand order repaired and checked by reading, not compiled, not run on hardware
//
#include <hip/hip_runtime.h>

// ---------------------------------------------------------------------------
// Problem constants (match reference)
// ---------------------------------------------------------------------------
#define BB 64
#define SS 2048
#define E2 512
#define DD 512
#define AA 512
#define CC 100
#define LN_EPS 1e-5f

// ---------------------------------------------------------------------------
// WMMA types / helpers
// ---------------------------------------------------------------------------
typedef __attribute__((ext_vector_type(16))) __bf16 v16bf;
typedef __attribute__((ext_vector_type(8)))  float  v8f;

union V16BF {
    v16bf v;
    unsigned short u[16];
    unsigned int   u32[8];
    uint4          q[2];
};

union BF2 { __bf16 h[2]; unsigned int u; };

// packed fp32x2 -> bf16x2 via compiler-native conversion (expect v_cvt_pk_bf16_f32)
static __device__ __forceinline__ unsigned int pk_bf16(float x, float y) {
    BF2 t;
    t.h[0] = (__bf16)x;
    t.h[1] = (__bf16)y;
    return t.u;
}

static __device__ __forceinline__ unsigned short bf16_bits(float x) {
    BF2 t;
    t.h[0] = (__bf16)x;
    t.h[1] = (__bf16)0.0f;
    return (unsigned short)(t.u & 0xffffu);
}

static __device__ __forceinline__ float sigm(float x) {
    return 1.0f / (1.0f + __expf(-x));
}

// ---------------------------------------------------------------------------
// Kernel 0: enc_W fp32 [A,E2] -> bf16 row-major (B-fragment friendly layout)
// ---------------------------------------------------------------------------
__global__ __launch_bounds__(256) void convert_w_kernel(
    const float* __restrict__ w, unsigned int* __restrict__ wb)
{
    const int idx = blockIdx.x * 256 + threadIdx.x;     // AA*E2/2 = 131072
    const float2 f = *(const float2*)(w + 2 * idx);
    wb[idx] = pk_bf16(f.x, f.y);
}

// ---------------------------------------------------------------------------
// Kernel 1: dec_proj[b,a] = hidden[b,:] . dec_W[a,:]
// ---------------------------------------------------------------------------
__global__ __launch_bounds__(256) void dec_proj_kernel(
    const float* __restrict__ hidden, const float* __restrict__ decW,
    float* __restrict__ dp)
{
    int idx = blockIdx.x * 256 + threadIdx.x;          // BB*AA = 32768
    if (idx >= BB * AA) return;
    int a = idx >> 6;
    int b = idx & 63;
    const float* h = hidden + b * DD;
    const float* w = decW + (size_t)a * DD;
    float s = 0.f;
    for (int k = 0; k < DD; ++k) s += h[k] * w[k];
    dp[b * AA + a] = s;
}

// ---------------------------------------------------------------------------
// Kernel 2 (hot): fused enc_proj GEMM (bf16 WMMA, f32 accum) + dec_proj add
//                 + LayerNorm + tanh + energy dot -> energy[B*S]
// One workgroup = 32 rows of (b,s) x all 512 'a' columns.
// 4 waves; wave w owns columns [w*128, w*128+128) = 8 tiles, x 2 row-groups.
// B fragments (enc_W bf16) are raw b128 loads; A fragments convert inline.
// ---------------------------------------------------------------------------
__global__ __launch_bounds__(128) void attn_energy_kernel(
    const float* __restrict__ enc,             // [B,S,E2] fp32
    const unsigned short* __restrict__ encWbf, // [A,E2] bf16
    const float* __restrict__ dec_proj,        // [B,A]
    const float* __restrict__ ln_g,
    const float* __restrict__ ln_b,
    const float* __restrict__ eW,              // [A]
    float* __restrict__ energy)                // [B*S]
{
    __shared__ float sm[32 * AA];                      // 64 KB

    const int row_base = blockIdx.x * 32;              // global row = b*S + s
    const int bidx     = row_base / SS;
    const int tid      = threadIdx.x;
    const int wave     = tid >> 5;
    const int lane     = tid & 31;
    const int half     = lane >> 4;                    // 0 or 1
    const int l16      = lane & 15;

    v8f acc[2][8];
    #pragma unroll
    for (int rg = 0; rg < 2; ++rg)
        #pragma unroll
        for (int t = 0; t < 8; ++t)
            #pragma unroll
            for (int r = 0; r < 8; ++r) acc[rg][t][r] = 0.0f;

    // A-matrix rows this lane feeds (both lane halves carry rows 0..15 of group)
    const float* arow0 = enc + (size_t)(row_base + l16) * E2;
    const float* arow1 = enc + (size_t)(row_base + 16 + l16) * E2;

    for (int kb = 0; kb < E2; kb += 32) {
        // ---- A fragments: 16x32 bf16, per-ISA layout -------------------
        // element e: K = (e>>3)*16 + half*8 + (e & 7); pairs are consecutive K
        V16BF a0, a1;
        #pragma unroll
        for (int v = 0; v < 8; ++v) {
            const int g  = v >> 2;
            const int j  = v & 3;
            const int k0 = kb + g * 16 + half * 8 + 2 * j;
            const float2 f0 = *(const float2*)(arow0 + k0);
            const float2 f1 = *(const float2*)(arow1 + k0);
            a0.u32[v] = pk_bf16(f0.x, f0.y);
            a1.u32[v] = pk_bf16(f1.x, f1.y);
        }
        // ---- 8 column tiles; B frag shared across both row groups -----
        #pragma unroll
        for (int nt = 0; nt < 8; ++nt) {
            const int col = wave * 128 + nt * 16 + l16;       // N = lane&15
            const unsigned short* bw =
                encWbf + (size_t)col * E2 + kb + half * 8;    // K = half*8+[0,8) and 16+half*8+[0,8)
            V16BF bfv;
            bfv.q[0] = *(const uint4*)(bw);
            bfv.q[1] = *(const uint4*)(bw + 16);
            acc[0][nt] = __builtin_amdgcn_wmma_f32_16x16x32_bf16(
                false, a0.v, false, bfv.v, (short)0, acc[0][nt], false, false);
            acc[1][nt] = __builtin_amdgcn_wmma_f32_16x16x32_bf16(
                false, a1.v, false, bfv.v, (short)0, acc[1][nt], false, false);
        }
    }

    // ---- add dec_proj (row-invariant over s) and spill tiles to LDS ----
    #pragma unroll
    for (int nt = 0; nt < 8; ++nt) {
        const int col = wave * 128 + nt * 16 + l16;
        const float dp = dec_proj[bidx * AA + col];
        #pragma unroll
        for (int rg = 0; rg < 2; ++rg)
            #pragma unroll
            for (int r = 0; r < 8; ++r) {
                const int m = rg * 16 + r + half * 8;  // C/D layout: M = r + 8*half
                sm[m * AA + col] = acc[rg][nt][r] + dp;
            }
    }
    __syncthreads();

    // ---- LayerNorm + tanh + energy dot; wave handles 8 rows -----------
    {
        const int row = wave * 8 + (lane >> 2);        // 4 lanes per row
        const int l4  = lane & 3;
        const float* rp = sm + row * AA + l4 * 128;
        float s = 0.f, ss = 0.f;
        for (int c = 0; c < 128; ++c) {
            const float v = rp[c];
            s += v; ss += v * v;
        }
        #pragma unroll
        for (int m = 1; m < 4; m <<= 1) {
            s  += __shfl_xor(s,  m, 4);
            ss += __shfl_xor(ss, m, 4);
        }
        const float mean = s * (1.0f / (float)AA);
        const float var  = ss * (1.0f / (float)AA) - mean * mean;
        const float rstd = rsqrtf(var + LN_EPS);

        float e = 0.f;
        const int cbase = l4 * 128;
        for (int c = 0; c < 128; ++c) {
            const float v = (rp[c] - mean) * rstd * ln_g[cbase + c] + ln_b[cbase + c];
            e += tanhf(v) * eW[cbase + c];
        }
        #pragma unroll
        for (int m = 1; m < 4; m <<= 1) e += __shfl_xor(e, m, 4);
        if (l4 == 0) energy[row_base + row] = e;
    }
}

// ---------------------------------------------------------------------------
// Kernel 3: masked softmax over S per batch row -> attn (written to d_out)
// ---------------------------------------------------------------------------
__global__ __launch_bounds__(256) void softmax_kernel(
    const float* __restrict__ energy, const unsigned char* __restrict__ mask,
    float* __restrict__ attn)
{
    __shared__ float red[256];
    const int b   = blockIdx.x;
    const int tid = threadIdx.x;

    float vals[8];
    float mx = -3.4e38f;
    #pragma unroll
    for (int i = 0; i < 8; ++i) {
        const int s = tid + i * 256;
        float e = energy[b * SS + s];
        if (!mask[b * SS + s]) e = -3.4e38f;
        vals[i] = e;
        mx = fmaxf(mx, e);
    }
    red[tid] = mx; __syncthreads();
    for (int off = 128; off > 0; off >>= 1) {
        if (tid < off) red[tid] = fmaxf(red[tid], red[tid + off]);
        __syncthreads();
    }
    mx = red[0]; __syncthreads();

    float sum = 0.f;
    #pragma unroll
    for (int i = 0; i < 8; ++i) {
        vals[i] = __expf(vals[i] - mx);
        sum += vals[i];
    }
    red[tid] = sum; __syncthreads();
    for (int off = 128; off > 0; off >>= 1) {
        if (tid < off) red[tid] += red[tid + off];
        __syncthreads();
    }
    const float inv = 1.0f / red[0];
    #pragma unroll
    for (int i = 0; i < 8; ++i)
        attn[b * SS + tid + i * 256] = vals[i] * inv;
}

// ---------------------------------------------------------------------------
// Kernel 4: context[b,e] = sum_s attn[b,s] * enc[b,s,e]   (coalesced over e)
// ---------------------------------------------------------------------------
__global__ __launch_bounds__(256) void context_kernel(
    const float* __restrict__ enc, const float* __restrict__ attn,
    float* __restrict__ ctx)
{
    const int b = blockIdx.x;
    const int e = blockIdx.y * 256 + threadIdx.x;
    const float* ep = enc + (size_t)b * SS * E2 + e;
    const float* ap = attn + b * SS;
    float s = 0.f;
    for (int t = 0; t < SS; ++t) s += ap[t] * ep[(size_t)t * E2];
    ctx[b * E2 + e] = s;
}

// ---------------------------------------------------------------------------
// Kernel 5: x = concat(emb[token], context)   [B, D+E2]
// ---------------------------------------------------------------------------
__global__ __launch_bounds__(256) void build_x_kernel(
    const int* __restrict__ tok, const float* __restrict__ emb,
    const float* __restrict__ ctx, float* __restrict__ x)
{
    const int idx = blockIdx.x * 256 + threadIdx.x;    // B*(D+E2) = 65536
    const int b = idx >> 10;
    const int k = idx & 1023;
    x[idx] = (k < DD) ? emb[(size_t)tok[b] * DD + k] : ctx[b * E2 + (k - DD)];
}

// ---------------------------------------------------------------------------
// Kernel 6: gates[b, 4D] = x @ W_ih^T + hidden @ W_hh^T + b_ih + b_hh
// ---------------------------------------------------------------------------
__global__ __launch_bounds__(256) void gates_kernel(
    const float* __restrict__ x, const float* __restrict__ hidden,
    const float* __restrict__ Wih, const float* __restrict__ Whh,
    const float* __restrict__ bih, const float* __restrict__ bhh,
    float* __restrict__ gates)
{
    const int idx = blockIdx.x * 256 + threadIdx.x;    // B*4D = 131072
    const int j = idx >> 6;
    const int b = idx & 63;
    const float* xr = x + b * (DD + E2);
    const float* wr = Wih + (size_t)j * (DD + E2);
    float s = bih[j] + bhh[j];
    for (int k = 0; k < DD + E2; ++k) s += xr[k] * wr[k];
    const float* hr = hidden + b * DD;
    const float* w2 = Whh + (size_t)j * DD;
    for (int k = 0; k < DD; ++k) s += hr[k] * w2[k];
    gates[b * 4 * DD + j] = s;
}

// ---------------------------------------------------------------------------
// Kernel 7: LSTM pointwise + build out_in = concat(new_hidden, context)
// ---------------------------------------------------------------------------
__global__ __launch_bounds__(256) void lstm_pointwise_kernel(
    const float* __restrict__ gates, const float* __restrict__ cell,
    const float* __restrict__ ctx,
    float* __restrict__ new_hidden, float* __restrict__ new_cell,
    float* __restrict__ out_in)
{
    const int idx = blockIdx.x * 256 + threadIdx.x;    // B*D = 32768
    const int b = idx >> 9;
    const int d = idx & 511;
    const float* g = gates + b * 4 * DD;
    const float i  = sigm(g[d]);
    const float f  = sigm(g[DD + d]);
    const float gg = tanhf(g[2 * DD + d]);
    const float o  = sigm(g[3 * DD + d]);
    const float nc = f * cell[idx] + i * gg;
    const float nh = o * tanhf(nc);
    new_cell[idx]   = nc;
    new_hidden[idx] = nh;
    out_in[b * (DD + E2) + d]      = nh;
    out_in[b * (DD + E2) + DD + d] = ctx[idx];
}

// ---------------------------------------------------------------------------
// Kernel 8: h1 = tanh(out_in @ fc1_W^T + fc1_b)
// ---------------------------------------------------------------------------
__global__ __launch_bounds__(256) void fc1_kernel(
    const float* __restrict__ out_in, const float* __restrict__ W,
    const float* __restrict__ bias, float* __restrict__ h1)
{
    const int idx = blockIdx.x * 256 + threadIdx.x;    // B*D = 32768
    const int j = idx >> 6;
    const int b = idx & 63;
    const float* xr = out_in + b * (DD + E2);
    const float* wr = W + (size_t)j * (DD + E2);
    float s = bias[j];
    for (int k = 0; k < DD + E2; ++k) s += xr[k] * wr[k];
    h1[b * DD + j] = tanhf(s);
}

// ---------------------------------------------------------------------------
// Kernel 9: output = h1 @ fc2_W^T + fc2_b
// ---------------------------------------------------------------------------
__global__ __launch_bounds__(256) void fc2_kernel(
    const float* __restrict__ h1, const float* __restrict__ W,
    const float* __restrict__ bias, float* __restrict__ outp)
{
    const int idx = blockIdx.x * 256 + threadIdx.x;    // B*C = 6400
    if (idx >= BB * CC) return;
    const int j = idx >> 6;                             // 0..99
    const int b = idx & 63;
    const float* xr = h1 + b * DD;
    const float* wr = W + (size_t)j * DD;
    float s = bias[j];
    for (int k = 0; k < DD; ++k) s += xr[k] * wr[k];
    outp[b * CC + j] = s;
}

// ---------------------------------------------------------------------------
// Launch
// ---------------------------------------------------------------------------
extern "C" void kernel_launch(void* const* d_in, const int* in_sizes, int n_in,
                              void* d_out, int out_size, void* d_ws, size_t ws_size,
                              hipStream_t stream) {
    (void)in_sizes; (void)n_in; (void)out_size; (void)ws_size;

    const int*   tok     = (const int*)  d_in[0];
    const float* hidden  = (const float*)d_in[1];
    const float* cell    = (const float*)d_in[2];
    const float* enc     = (const float*)d_in[3];
    const unsigned char* mask = (const unsigned char*)d_in[4];
    const float* emb     = (const float*)d_in[5];
    const float* encW    = (const float*)d_in[6];
    const float* decW    = (const float*)d_in[7];
    const float* ln_g    = (const float*)d_in[8];
    const float* ln_b    = (const float*)d_in[9];
    const float* energyW = (const float*)d_in[10];
    const float* W_ih    = (const float*)d_in[11];
    const float* W_hh    = (const float*)d_in[12];
    const float* b_ih    = (const float*)d_in[13];
    const float* b_hh    = (const float*)d_in[14];
    const float* fc1_W   = (const float*)d_in[15];
    const float* fc1_b   = (const float*)d_in[16];
    const float* fc2_W   = (const float*)d_in[17];
    const float* fc2_b   = (const float*)d_in[18];

    // d_out layout: output [B,C] | new_hidden [B,D] | new_cell [B,D] | attn [B,S]
    float* out_output = (float*)d_out;
    float* out_nh     = out_output + BB * CC;
    float* out_nc     = out_nh + BB * DD;
    float* out_attn   = out_nc + BB * DD;

    // workspace (byte offsets)
    char* wsb = (char*)d_ws;
    unsigned short* encWbf = (unsigned short*)(wsb);              // 512 KB
    float* dec_proj = (float*)(wsb + (512u << 10));               // 128 KB
    float* energy   = (float*)(wsb + (640u << 10));               // 512 KB
    float* context  = (float*)(wsb + (1152u << 10));              // 128 KB
    float* x        = (float*)(wsb + (1280u << 10));              // 256 KB
    float* gates    = (float*)(wsb + (1536u << 10));              // 512 KB
    float* out_in   = (float*)(wsb + (2048u << 10));              // 256 KB

    convert_w_kernel<<<(AA * E2 / 2) / 256, 256, 0, stream>>>(
        encW, (unsigned int*)encWbf);

    dec_proj_kernel<<<(BB * AA + 255) / 256, 256, 0, stream>>>(hidden, decW, dec_proj);

    attn_energy_kernel<<<(BB * SS) / 32, 128, 0, stream>>>(
        enc, encWbf, dec_proj, ln_g, ln_b, energyW, energy);

    softmax_kernel<<<BB, 256, 0, stream>>>(energy, mask, out_attn);

    context_kernel<<<dim3(BB, E2 / 256), 256, 0, stream>>>(enc, out_attn, context);

    build_x_kernel<<<(BB * (DD + E2)) / 256, 256, 0, stream>>>(tok, emb, context, x);

    gates_kernel<<<(BB * 4 * DD) / 256, 256, 0, stream>>>(
        x, hidden, W_ih, W_hh, b_ih, b_hh, gates);

    lstm_pointwise_kernel<<<(BB * DD) / 256, 256, 0, stream>>>(
        gates, cell, context, out_nh, out_nc, out_in);

    fc1_kernel<<<(BB * DD) / 256, 256, 0, stream>>>(out_in, fc1_W, fc1_b, x /*reuse*/);
    fc2_kernel<<<(BB * CC + 255) / 256, 256, 0, stream>>>(x, fc2_W, fc2_b, out_output);
}
